// GraphTransformer_72473278152877
// MI455X (gfx1250) — matrix operand order repaired and checked by reading, not compiled, not run on hardware
//
#include <hip/hip_runtime.h>

// ---------------------------------------------------------------------------
// GraphTransformer forward for MI455X (gfx1250, wave32, WMMA + TDM).
// B=2, N=256, IN=128, E_IN=32, H=256, DEPTH=4, HEADS=8, DH=32
// ---------------------------------------------------------------------------

typedef __attribute__((ext_vector_type(16))) __bf16 v16bf;
typedef __attribute__((ext_vector_type(8)))  float  v8f;
typedef __attribute__((ext_vector_type(2)))  float  v2f;
typedef __attribute__((ext_vector_type(4)))  unsigned int v4u;
typedef __attribute__((ext_vector_type(8)))  int    v8i;
typedef __attribute__((ext_vector_type(4)))  int    v4i;

#define BB 2
#define NN 256
#define IN_DIM 128
#define E_IN 32
#define HD 256
#define DEPTH 4
#define HEADS 8
#define DHH 32

#define NEG_MAX (-3.4028234663852886e38f)
#define ATT_SCALE 0.17677669529663687f  // 32^-0.5

// ---------------------------------------------------------------------------
// Exact-precision GEMM: C[M,N] = A[M,K] @ W[K,N] + bias, via v_wmma_f32_16x16x4_f32.
// One wave per 16x16 output tile; 8 waves per block.
// ---------------------------------------------------------------------------
__global__ void gemm_f32_wmma(const float* __restrict__ A, const float* __restrict__ W,
                              const float* __restrict__ bias, float* __restrict__ C,
                              int M, int K, int N) {
  int wave = blockIdx.x * 8 + (threadIdx.x >> 5);
  int lane = threadIdx.x & 31;
  int ntile = N >> 4;
  int tm = wave / ntile, tn = wave % ntile;
  int row0 = tm << 4;
  if (row0 >= M) return;              // wave-uniform; EXEC stays all-ones for WMMA
  int col0 = tn << 4;
  int hv = lane >> 4;                 // lane half
  int l15 = lane & 15;
  int arow = row0 + l15;
  int bcol = col0 + l15;

  v8f acc = {0.f, 0.f, 0.f, 0.f, 0.f, 0.f, 0.f, 0.f};
  const float* ap = A + (size_t)arow * K;
  for (int k0 = 0; k0 < K; k0 += 4) {
    v2f a, b;
    a[0] = ap[k0 + hv * 2 + 0];
    a[1] = ap[k0 + hv * 2 + 1];
    b[0] = W[(size_t)(k0 + hv * 2 + 0) * N + bcol];
    b[1] = W[(size_t)(k0 + hv * 2 + 1) * N + bcol];
    acc = __builtin_amdgcn_wmma_f32_16x16x4_f32(false, a, false, b, (short)0, acc,
                                                false, false);
  }
#pragma unroll
  for (int r = 0; r < 8; ++r) {
    int m = row0 + hv * 8 + r;
    C[(size_t)m * N + bcol] = acc[r] + bias[bcol];
  }
}

// ---------------------------------------------------------------------------
// f32 -> bf16 elementwise conversion (stages e / ek_w for the TDM+WMMA GEMM).
// ---------------------------------------------------------------------------
__global__ void f32_to_bf16_kernel(const float* __restrict__ src,
                                   __bf16* __restrict__ dst, int n) {
  int i = blockIdx.x * 256 + threadIdx.x;
  if (i < n) dst[i] = (__bf16)src[i];
}

// ---------------------------------------------------------------------------
// TDM 2-D tile load descriptor (D# groups per CDNA5 ISA).
// data_size = 2 bytes (bf16). Groups 2/3 zero (2-D tensor).
// 6-arg builtin form (clang-23 / therock-10.0 headers).
// ---------------------------------------------------------------------------
__device__ __forceinline__ void tdm_load_2d_bf16(unsigned lds_off, const void* gaddr,
                                                 unsigned tensor_d0, unsigned tensor_d1,
                                                 unsigned tile_d0, unsigned tile_d1,
                                                 unsigned stride0) {
  unsigned long long ga = (unsigned long long)gaddr;
  v4u g0;
  g0[0] = 1u;                                                  // count=1 valid descriptor
  g0[1] = lds_off;                                             // LDS byte address
  g0[2] = (unsigned)ga;                                        // global addr [31:0]
  g0[3] = (unsigned)((ga >> 32) & 0x01FFFFFFu) | (2u << 30);   // addr[56:32] | type=2
  v8i g1;
  g1[0] = (int)(1u << 16);                                     // wg_mask=0, data_size=1 (2B)
  g1[1] = (int)((tensor_d0 & 0xFFFFu) << 16);                  // bar_addr=0 | dim0 lo16
  g1[2] = (int)((tensor_d0 >> 16) | ((tensor_d1 & 0xFFFFu) << 16));
  g1[3] = (int)((tensor_d1 >> 16) | (tile_d0 << 16));
  g1[4] = (int)(tile_d1 & 0xFFFFu);                            // tile_dim1 | tile_dim2=0
  g1[5] = (int)stride0;                                        // dim0 stride lo32
  g1[6] = 0;                                                   // stride0 hi | stride1 lo
  g1[7] = 0;
  v4i z4 = {0, 0, 0, 0};
  v8i z8 = {0, 0, 0, 0, 0, 0, 0, 0};
  __builtin_amdgcn_tensor_load_to_lds(g0, g1, z4, z4, z8, 0);
}

// ---------------------------------------------------------------------------
// Dominant GEMM: ekv = e_bf16[131072,256] @ ekw_bf16[256,256] + bias -> bf16.
// Block = 256 threads (8 waves), 64x64 macro tile, whole K=256 resident in LDS.
// A slab (64x256 bf16, 32KB) and W slab (256x64 bf16, 32KB) arrive via
// tensor_load_to_lds; wave 0 waits on TENSORcnt, then __syncthreads().
// Each wave owns one 16-col strip: B fragments cached in registers (8 x v16bf),
// A fragments streamed from LDS (2 contiguous 16B reads per K-step).
// ---------------------------------------------------------------------------
__global__ void __launch_bounds__(256)
ekv_gemm_tdm(const __bf16* __restrict__ Ab, const __bf16* __restrict__ Wb,
             const float* __restrict__ bias, __bf16* __restrict__ Cb) {
  extern __shared__ __bf16 smem[];
  __bf16* Alds = smem;             // [64][256]
  __bf16* Wlds = smem + 64 * 256;  // [256][64]  (k-major: Wlds[k*64 + c])

  int bidN = blockIdx.x & 3;       // 4 col blocks of 64
  int bidM = blockIdx.x >> 2;      // 2048 row blocks of 64
  int row0 = bidM * 64;
  int col0 = bidN * 64;

  if (threadIdx.x < 32) {          // wave 0 issues both DMAs (EXEC ignored by TDM)
    unsigned aoff = (unsigned)(size_t)(void*)Alds;   // flat LDS addr -> low 32 = LDS offset
    unsigned woff = (unsigned)(size_t)(void*)Wlds;
    tdm_load_2d_bf16(aoff, Ab + (size_t)row0 * HD, HD, (unsigned)(BB * NN * NN),
                     HD, 64, HD);
    tdm_load_2d_bf16(woff, Wb + col0, HD, HD, 64, HD, HD);
    __builtin_amdgcn_s_wait_tensorcnt(0);
  }
  __syncthreads();

  int lane = threadIdx.x & 31;
  int wv = threadIdx.x >> 5;       // 0..7
  int hv = lane >> 4;
  int l15 = lane & 15;
  int colc = (wv & 3) * 16 + l15;  // column within macro tile

  v16bf bfr[8];                    // full K chain of B fragments, register resident
#pragma unroll
  for (int ks = 0; ks < 8; ++ks) {
    const __bf16* wp = Wlds + (ks * 32 + hv * 16) * 64 + colc;
#pragma unroll
    for (int e = 0; e < 16; ++e) bfr[ks][e] = wp[e * 64];
  }

#pragma unroll
  for (int t = 0; t < 2; ++t) {
    int mt = (wv >> 2) * 2 + t;    // 4 row tiles split across wave pairs
    int arow = mt * 16 + l15;
    v8f acc = {0.f, 0.f, 0.f, 0.f, 0.f, 0.f, 0.f, 0.f};
#pragma unroll
    for (int ks = 0; ks < 8; ++ks) {
      const __bf16* ap = Alds + arow * HD + ks * 32 + hv * 16;
      v16bf a;
#pragma unroll
      for (int e = 0; e < 8; ++e) a[e] = ap[e];        // K = ks*32+hv*16 + 0..7
#pragma unroll
      for (int e = 0; e < 8; ++e) a[8 + e] = ap[8 + e];  // K += 8
      acc = __builtin_amdgcn_wmma_f32_16x16x32_bf16(false, a, false, bfr[ks],
                                                    (short)0, acc, false, false);
    }
#pragma unroll
    for (int r = 0; r < 8; ++r) {
      int m = row0 + mt * 16 + hv * 8 + r;
      int c = col0 + colc;
      Cb[(size_t)m * HD + c] = (__bf16)(acc[r] + bias[c]);
    }
  }
}

// ---------------------------------------------------------------------------
// LayerNorm over last dim (256). One block per row, LDS tree reduction.
// ---------------------------------------------------------------------------
__global__ void ln_kernel(const float* __restrict__ x, const float* __restrict__ g,
                          const float* __restrict__ bta, float* __restrict__ xn) {
  __shared__ float red[HD];
  int row = blockIdx.x;
  int t = threadIdx.x;
  float v = x[(size_t)row * HD + t];
  red[t] = v;
  __syncthreads();
  for (int s = HD / 2; s > 0; s >>= 1) {
    if (t < s) red[t] += red[t + s];
    __syncthreads();
  }
  float mu = red[0] * (1.0f / HD);
  __syncthreads();
  float dv = v - mu;
  red[t] = dv * dv;
  __syncthreads();
  for (int s = HD / 2; s > 0; s >>= 1) {
    if (t < s) red[t] += red[t + s];
    __syncthreads();
  }
  float var = red[0] * (1.0f / HD);
  xn[(size_t)row * HD + t] = g[t] * dv * rsqrtf(var + 1e-5f) + bta[t];
}

// ---------------------------------------------------------------------------
// Fused sim + masked softmax. One wave per (b, h, i); lanes stripe over j.
// Reference quirk: VALID key positions (j < len) are filled with -FLT_MAX.
// ---------------------------------------------------------------------------
__global__ void attn_sim_kernel(const float* __restrict__ q, const float* __restrict__ kv,
                                const __bf16* __restrict__ ekv,
                                const int* __restrict__ lengths,
                                float* __restrict__ attn) {
  int wid = blockIdx.x * 8 + (threadIdx.x >> 5);  // ((b*HEADS + h)*NN + i)
  int lane = threadIdx.x & 31;
  int i = wid & (NN - 1);
  int h = (wid >> 8) & (HEADS - 1);
  int b = wid >> 11;

  float qreg[DHH];
  const float* qp = q + ((size_t)(b * NN + i)) * HD + h * DHH;
#pragma unroll
  for (int d = 0; d < DHH; ++d) qreg[d] = qp[d];

  int len = lengths[b];
  float s[8];
  float mx = NEG_MAX;
#pragma unroll
  for (int jj = 0; jj < 8; ++jj) {
    int j = lane + jj * 32;
    const float* kp = kv + ((size_t)(b * NN + j)) * (2 * HD) + h * DHH;
    const __bf16* ep = ekv + (((size_t)(b * NN + i) * NN + j)) * HD + h * DHH;
    float acc = 0.f;
#pragma unroll
    for (int d = 0; d < DHH; ++d) acc = fmaf(qreg[d], kp[d] + (float)ep[d], acc);
    float sv = (j < len) ? NEG_MAX : acc * ATT_SCALE;
    s[jj] = sv;
    mx = fmaxf(mx, sv);
  }
#pragma unroll
  for (int off = 16; off > 0; off >>= 1) mx = fmaxf(mx, __shfl_xor(mx, off, 32));
  float sum = 0.f;
#pragma unroll
  for (int jj = 0; jj < 8; ++jj) {
    s[jj] = __expf(s[jj] - mx);  // all-(-max) rows -> exp(0)=1 -> uniform, as in jax
    sum += s[jj];
  }
#pragma unroll
  for (int off = 16; off > 0; off >>= 1) sum += __shfl_xor(sum, off, 32);
  float inv = 1.f / sum;
  float* ar = attn + (size_t)wid * NN;
#pragma unroll
  for (int jj = 0; jj < 8; ++jj) ar[lane + jj * 32] = s[jj] * inv;
}

// ---------------------------------------------------------------------------
// out[i,dh] = sum_j attn[i,j] * (v[j,dh] + ekv[i,j,dh]). One wave per (b,h,i),
// lane = dh. Written as [b, i, h*DH+dh] so the o-projection is a plain GEMM.
// ---------------------------------------------------------------------------
__global__ void attn_out_kernel(const float* __restrict__ attn, const float* __restrict__ kv,
                                const __bf16* __restrict__ ekv, float* __restrict__ out) {
  int wid = blockIdx.x * 8 + (threadIdx.x >> 5);
  int lane = threadIdx.x & 31;
  int i = wid & (NN - 1);
  int h = (wid >> 8) & (HEADS - 1);
  int b = wid >> 11;

  const float* ar = attn + (size_t)wid * NN;
  float acc = 0.f;
  for (int j = 0; j < NN; ++j) {
    float a = ar[j];
    float vv = kv[((size_t)(b * NN + j)) * (2 * HD) + HD + h * DHH + lane];
    float ee = (float)ekv[(((size_t)(b * NN + i) * NN + j)) * HD + h * DHH + lane];
    acc = fmaf(a, vv + ee, acc);
  }
  out[((size_t)(b * NN + i)) * HD + h * DHH + lane] = acc;
}

// ---------------------------------------------------------------------------
// Gate + residual: g = sigmoid([o, x, o-x] . gw);  x = o*g + x*(1-g) in place.
// ---------------------------------------------------------------------------
__global__ void gate_kernel(const float* __restrict__ out2, float* __restrict__ x,
                            const float* __restrict__ gw) {
  int wid = blockIdx.x * 8 + (threadIdx.x >> 5);  // b*NN + n
  int lane = threadIdx.x & 31;
  const float* op = out2 + (size_t)wid * HD;
  float* xp = x + (size_t)wid * HD;
  float acc = 0.f;
#pragma unroll
  for (int e = 0; e < 8; ++e) {
    int c = lane + e * 32;
    float o = op[c], xv = xp[c];
    acc += o * gw[c] + xv * gw[HD + c] + (o - xv) * gw[2 * HD + c];
  }
#pragma unroll
  for (int off = 16; off > 0; off >>= 1) acc += __shfl_xor(acc, off, 32);
  float g = 1.f / (1.f + __expf(-acc));
#pragma unroll
  for (int e = 0; e < 8; ++e) {
    int c = lane + e * 32;
    xp[c] = op[c] * g + xp[c] * (1.f - g);
  }
}

// ---------------------------------------------------------------------------
extern "C" void kernel_launch(void* const* d_in, const int* in_sizes, int n_in,
                              void* d_out, int out_size, void* d_ws, size_t ws_size,
                              hipStream_t stream) {
  const float* nodes   = (const float*)d_in[0];
  const float* edges   = (const float*)d_in[1];
  const int*   lengths = (const int*)d_in[2];
  const float* n_emb_w = (const float*)d_in[3];
  const float* n_emb_b = (const float*)d_in[4];
  const float* e_emb_w = (const float*)d_in[5];
  const float* e_emb_b = (const float*)d_in[6];
  const float* ln_g    = (const float*)d_in[7];
  const float* ln_b    = (const float*)d_in[8];
  const float* q_w     = (const float*)d_in[9];
  const float* q_b     = (const float*)d_in[10];
  const float* kv_w    = (const float*)d_in[11];
  const float* kv_b    = (const float*)d_in[12];
  const float* ek_w    = (const float*)d_in[13];
  const float* ek_b    = (const float*)d_in[14];
  const float* o_w     = (const float*)d_in[15];
  const float* o_b     = (const float*)d_in[16];
  const float* gate_w  = (const float*)d_in[17];

  float* x = (float*)d_out;                        // [B,N,H]
  float* e = (float*)d_out + (size_t)BB * NN * HD; // [B,N,N,H] output region

  const int MROW = BB * NN;        // 512
  const int MEDG = BB * NN * NN;   // 131072

  // workspace layout
  char* ws = (char*)d_ws;
  size_t off = 0;
  float*  xn   = (float*)(ws + off); off += (size_t)MROW * HD * 4;        // 512 KB
  float*  qbuf = (float*)(ws + off); off += (size_t)MROW * HD * 4;        // 512 KB
  float*  kvb  = (float*)(ws + off); off += (size_t)MROW * 2 * HD * 4;    // 1 MB
  float*  aout = (float*)(ws + off); off += (size_t)MROW * HD * 4;        // 512 KB
  float*  out2 = (float*)(ws + off); off += (size_t)MROW * HD * 4;        // 512 KB
  float*  attn = (float*)(ws + off); off += (size_t)BB * HEADS * NN * NN * 4;  // 4 MB
  __bf16* ekv  = (__bf16*)(ws + off); off += (size_t)MEDG * HD * 2;       // 64 MB
  __bf16* e_bf = (__bf16*)(ws + off); off += (size_t)MEDG * HD * 2;       // 64 MB
  __bf16* ekwb = (__bf16*)(ws + off); off += (size_t)DEPTH * HD * HD * 2; // 512 KB

  // embeddings (exact f32 WMMA)
  gemm_f32_wmma<<<(MROW / 16) * (HD / 16) / 8, 256, 0, stream>>>(
      nodes, n_emb_w, n_emb_b, x, MROW, IN_DIM, HD);
  gemm_f32_wmma<<<(MEDG / 16) * (HD / 16) / 8, 256, 0, stream>>>(
      edges, e_emb_w, e_emb_b, e, MEDG, E_IN, HD);

  // stage bf16 copies for the TDM-fed ekv GEMM
  f32_to_bf16_kernel<<<(MEDG * HD) / 256, 256, 0, stream>>>(e, e_bf, MEDG * HD);
  f32_to_bf16_kernel<<<(DEPTH * HD * HD) / 256, 256, 0, stream>>>(
      ek_w, ekwb, DEPTH * HD * HD);

  for (int d = 0; d < DEPTH; ++d) {
    ln_kernel<<<MROW, HD, 0, stream>>>(x, ln_g + d * HD, ln_b + d * HD, xn);
    gemm_f32_wmma<<<(MROW / 16) * (HD / 16) / 8, 256, 0, stream>>>(
        xn, q_w + (size_t)d * HD * HD, q_b + d * HD, qbuf, MROW, HD, HD);
    gemm_f32_wmma<<<(MROW / 16) * (2 * HD / 16) / 8, 256, 0, stream>>>(
        xn, kv_w + (size_t)d * HD * 2 * HD, kv_b + d * 2 * HD, kvb, MROW, HD, 2 * HD);
    ekv_gemm_tdm<<<(MEDG / 64) * (HD / 64), 256, 65536, stream>>>(
        e_bf, ekwb + (size_t)d * HD * HD, ek_b + d * HD, ekv);
    attn_sim_kernel<<<BB * HEADS * NN / 8, 256, 0, stream>>>(qbuf, kvb, ekv, lengths, attn);
    attn_out_kernel<<<BB * HEADS * NN / 8, 256, 0, stream>>>(attn, kvb, ekv, aout);
    gemm_f32_wmma<<<(MROW / 16) * (HD / 16) / 8, 256, 0, stream>>>(
        aout, o_w + (size_t)d * HD * HD, o_b + d * HD, out2, MROW, HD, HD);
    gate_kernel<<<MROW / 8, 256, 0, stream>>>(out2, x, gate_w + d * 3 * HD);
  }
}
